// DeltaNet_31877247271469
// MI455X (gfx1250) — matrix operand order repaired and checked by reading, not compiled, not run on hardware
//
#include <hip/hip_runtime.h>

// Problem constants (from reference setup_inputs):
//   b=4, h=4, L=4096, dk=256, dv=256, chunk=32
#define CC     32            // chunk size
#define DK     256           // key dim
#define DVT    256           // total value dim
#define DVS    32            // dv slice handled per workgroup
#define NSLICE (DVT / DVS)   // 8 slices
#define LSEQ   4096
#define NCH    (LSEQ / CC)   // 128 chunks

typedef __attribute__((ext_vector_type(16))) __bf16       v16bf;
typedef __attribute__((ext_vector_type(8)))  float        v8f;
typedef __attribute__((ext_vector_type(8)))  unsigned int v8u;

static __device__ __forceinline__ unsigned short f2bf(float f) {
  unsigned u = __builtin_bit_cast(unsigned, f);
  u += 0x7FFFu + ((u >> 16) & 1u);   // round-to-nearest-even
  return (unsigned short)(u >> 16);
}

static __device__ __forceinline__ v8f vzero8() {
  v8f z = {0.f, 0.f, 0.f, 0.f, 0.f, 0.f, 0.f, 0.f};
  return z;
}

static __device__ __forceinline__ v8f wmma_bf16(v16bf a, v16bf b, v8f c) {
  return __builtin_amdgcn_wmma_f32_16x16x32_bf16(false, a, false, b, (short)0, c,
                                                 false, false);
}

// A fragment (16x32 bf16, M x K) from row-major u16 matrix, tile at (row0,col0).
// Lane L: m = L%16, K-base = 8*(L/16); VGPR v<4 -> K=2v.., v>=4 -> K=16+2(v-4)..
static __device__ __forceinline__ v16bf load_a(const unsigned short* M, int ld,
                                               int row0, int col0, int lane) {
  int g = lane >> 4, m = lane & 15;
  const unsigned short* p = M + (row0 + m) * ld + col0 + 8 * g;
  v8u r;
  r[0] = *(const unsigned int*)(p + 0);
  r[1] = *(const unsigned int*)(p + 2);
  r[2] = *(const unsigned int*)(p + 4);
  r[3] = *(const unsigned int*)(p + 6);
  r[4] = *(const unsigned int*)(p + 16);
  r[5] = *(const unsigned int*)(p + 18);
  r[6] = *(const unsigned int*)(p + 20);
  r[7] = *(const unsigned int*)(p + 22);
  return __builtin_bit_cast(v16bf, r);
}

// B fragment (32x16 bf16, K x N) from row-major u16 matrix, tile at (k0,col0).
// Lane L: n = L%16, K-base = 8*(L/16); VGPR v<4 -> K=2v.., v>=4 -> K=16+2(v-4)..
static __device__ __forceinline__ v16bf load_b(const unsigned short* M, int ld,
                                               int k0, int col0, int lane) {
  int g = lane >> 4, n = lane & 15;
  const unsigned short* p = M + (k0 + 8 * g) * ld + col0 + n;
  v8u r;
#pragma unroll
  for (int v = 0; v < 8; ++v) {
    int kb = 2 * v + (v >= 4 ? 8 : 0);
    r[v] = (unsigned int)p[kb * ld] | ((unsigned int)p[(kb + 1) * ld] << 16);
  }
  return __builtin_bit_cast(v16bf, r);
}

// C/D f32 fragment from row-major float matrix, tile at (row0,col0).
static __device__ __forceinline__ v8f load_c(const float* M, int ld, int row0,
                                             int col0, int lane) {
  int mh = (lane >> 4) * 8, n = lane & 15;
  v8f c;
#pragma unroll
  for (int r = 0; r < 8; ++r) c[r] = M[(row0 + mh + r) * ld + col0 + n];
  return c;
}

static __device__ __forceinline__ float bcast_lane(float x, int srcLane) {
  return __builtin_bit_cast(
      float, __builtin_amdgcn_readlane(__builtin_bit_cast(int, x), srcLane));
}

__global__ __launch_bounds__(256) void deltanet_kernel(
    const float* __restrict__ q, const float* __restrict__ k,
    const float* __restrict__ v, const float* __restrict__ beta,
    const float* __restrict__ gamma, float* __restrict__ out,
    float* __restrict__ outS) {
  // ---- LDS state & staging (~131 KB) ----
  __shared__ __align__(16) float          shS [DK * DVS];   // carried state (f32)
  __shared__ __align__(16) unsigned short shSb[DK * DVS];   // bf16 shadow of S
  __shared__ __align__(16) unsigned short shQ [CC * DK];    // l2norm(q), bf16
  __shared__ __align__(16) unsigned short shKt[DK * CC];    // l2norm(k)^T, bf16
  __shared__ __align__(16) unsigned short shKB[CC * DK];    // k*beta, bf16
  __shared__ __align__(16) unsigned short shW [CC * DK];    // -(T @ k_beta), bf16
  __shared__ __align__(16) unsigned short shV [CC * DVS];   // v*beta slice, bf16
  __shared__ __align__(16) unsigned short shT [CC * CC];    // T, bf16
  __shared__ __align__(16) unsigned short shAt[CC * CC];    // attn_local, bf16
  __shared__ __align__(16) unsigned short shU [CC * DVS];   // u_i, bf16
  __shared__ __align__(16) float          shAf[CC * CC];    // A (f32)
  __shared__ __align__(16) float          shUf[CC * DVS];   // u_raw (f32)
  __shared__ float shRed[CC][16];
  __shared__ float shGm;

  const int tid  = threadIdx.x;
  const int lane = tid & 31;
  const int wv   = tid >> 5;
  const int bh   = blockIdx.x / NSLICE;
  const int sl   = blockIdx.x % NSLICE;
  const size_t seq_base = (size_t)bh * LSEQ;   // row offset of this head

  // zero the carried state
  for (int i = tid; i < DK * DVS; i += 256) { shS[i] = 0.f; shSb[i] = 0; }
  __syncthreads();

  for (int ch = 0; ch < NCH; ++ch) {
    const int l0 = ch * CC;

    // ================= stage A: load + normalize + beta-scale =================
    {
      int row = tid >> 3, part = tid & 7;                 // 32 rows x 8 threads
      const size_t rbase = (seq_base + l0 + row) * (size_t)DK + part * 32;
      const size_t vbase =
          (seq_base + l0 + row) * (size_t)DVT + (size_t)sl * DVS + part * 4;
      float bta = beta[seq_base + l0 + row];
      float tv[32];
      float s;

      // ---- q pass ----
      s = 0.f;
#pragma unroll
      for (int i = 0; i < 32; ++i) { tv[i] = q[rbase + i]; s += tv[i] * tv[i]; }
      shRed[row][part] = s;

      // v*beta slice
#pragma unroll
      for (int i = 0; i < 4; ++i)
        shV[row * DVS + part * 4 + i] = f2bf(v[vbase + i] * bta);

      if (tid == 0) {
        float gs = 0.f;
        for (int i = 0; i < CC; ++i) gs += gamma[seq_base + l0 + i];
        shGm = gs * (1.0f / CC);
      }

      // prefetch next chunk while this one computes (global_prefetch_b8)
      if (ch + 1 < NCH) {
        __builtin_prefetch(q + rbase + (size_t)CC * DK, 0, 1);
        __builtin_prefetch(k + rbase + (size_t)CC * DK, 0, 1);
        __builtin_prefetch(v + vbase + (size_t)CC * DVT, 0, 1);
      }
      __syncthreads();

      float tq = 0.f;
#pragma unroll
      for (int i = 0; i < 8; ++i) tq += shRed[row][i];
      float rq = rsqrtf(tq + 1e-6f);
      unsigned short* qd = &shQ[row * DK + part * 32];
#pragma unroll
      for (int i = 0; i < 32; ++i) qd[i] = f2bf(tv[i] * rq);

      // ---- k pass (reuse tv registers) ----
      s = 0.f;
#pragma unroll
      for (int i = 0; i < 32; ++i) { tv[i] = k[rbase + i]; s += tv[i] * tv[i]; }
      shRed[row][8 + part] = s;
      __syncthreads();

      float tk = 0.f;
#pragma unroll
      for (int i = 0; i < 8; ++i) tk += shRed[row][8 + i];
      float rk = rsqrtf(tk + 1e-6f);
      unsigned short* kbd = &shKB[row * DK + part * 32];
      unsigned short* ktd = &shKt[(part * 32) * CC + row];
#pragma unroll
      for (int i = 0; i < 32; ++i) {
        float kn = tv[i] * rk;
        ktd[i * CC] = f2bf(kn);           // transposed copy k^T[dk][c]
        kbd[i]      = f2bf(kn * bta);
      }
    }
    __syncthreads();

    // ========= stage B: A = -(k_beta k^T) strict-lower ; attn = tril(q k^T) ====
    {
      int wq = wv & 3;
      int ti = wq >> 1, tj = wq & 1;
      const unsigned short* Am = (wv < 4) ? shKB : shQ;
      v8f acc = vzero8();
#pragma unroll
      for (int kk = 0; kk < DK; kk += 32) {
        v16bf a = load_a(Am, DK, ti * 16, kk, lane);
        v16bf b = load_b(shKt, CC, kk, tj * 16, lane);   // B = k^T (pre-stored)
        acc = wmma_bf16(a, b, acc);
      }
      int mh = (lane >> 4) * 8, n = lane & 15;
#pragma unroll
      for (int r = 0; r < 8; ++r) {
        int i = ti * 16 + mh + r, j = tj * 16 + n;
        if (wv < 4) shAf[i * CC + j] = (j >= i) ? 0.f : -acc[r];
        else        shAt[i * CC + j] = f2bf((i >= j) ? acc[r] : 0.f);
      }
    }
    __syncthreads();

    // ============== stage C: forward substitution -> T = (I-A)^-1 =============
    // Column-per-lane in registers; row broadcasts via v_readlane.
    if (wv == 0) {
      float a[CC];
#pragma unroll
      for (int r = 0; r < CC; ++r) a[r] = shAf[r * CC + lane];
#pragma unroll
      for (int i = 1; i < CC; ++i) {
        float inc = 0.f;
#pragma unroll
        for (int j = 0; j < i; ++j) inc += bcast_lane(a[i], j) * a[j];
        a[i] += (lane < i) ? inc : 0.f;
      }
#pragma unroll
      for (int r = 0; r < CC; ++r)
        shT[r * CC + lane] = f2bf(a[r] + (r == lane ? 1.f : 0.f));
    }
    __syncthreads();

    // =============== stage D: u_raw = T v_beta ; w = -(T k_beta) ==============
    for (int t = wv; t < 36; t += 8) {
      v8f acc = vzero8();
      int mh = (lane >> 4) * 8, n = lane & 15;
      if (t < 4) {
        int ti = t >> 1, tj = t & 1;
        v16bf a = load_a(shT, CC, ti * 16, 0, lane);
        v16bf b = load_b(shV, DVS, 0, tj * 16, lane);
        acc = wmma_bf16(a, b, acc);
#pragma unroll
        for (int r = 0; r < 8; ++r)
          shUf[(ti * 16 + mh + r) * DVS + tj * 16 + n] = acc[r];
      } else {
        int tt = t - 4;
        int ni = tt >> 4, nj = tt & 15;
        v16bf a = load_a(shT, CC, ni * 16, 0, lane);
        v16bf b = load_b(shKB, DK, 0, nj * 16, lane);
        acc = wmma_bf16(a, b, acc);
#pragma unroll
        for (int r = 0; r < 8; ++r)
          shW[(ni * 16 + mh + r) * DK + nj * 16 + n] = f2bf(-acc[r]);
      }
    }
    __syncthreads();

    // ====== stage E: u_i = u_raw + (-w) S (waves 0-3) ; o_acc = q S (4-7) =====
    v8f oacc = vzero8();
    if (wv < 4) {
      int ti = wv >> 1, tj = wv & 1;
      v8f acc = load_c(shUf, DVS, ti * 16, tj * 16, lane);
#pragma unroll
      for (int kk = 0; kk < DK; kk += 32) {
        v16bf a = load_a(shW, DK, ti * 16, kk, lane);
        v16bf b = load_b(shSb, DVS, kk, tj * 16, lane);
        acc = wmma_bf16(a, b, acc);
      }
      int mh = (lane >> 4) * 8, n = lane & 15;
#pragma unroll
      for (int r = 0; r < 8; ++r)
        shU[(ti * 16 + mh + r) * DVS + tj * 16 + n] = f2bf(acc[r]);
    } else {
      int ti = (wv - 4) >> 1, tj = (wv - 4) & 1;
#pragma unroll
      for (int kk = 0; kk < DK; kk += 32) {
        v16bf a = load_a(shQ, DK, ti * 16, kk, lane);
        v16bf b = load_b(shSb, DVS, kk, tj * 16, lane);
        oacc = wmma_bf16(a, b, oacc);
      }
    }
    __syncthreads();

    // ==== stage F: o = o_acc + attn u_i -> global (4-7) ; S *= gmean (0-3) ====
    if (wv >= 4) {
      int ti = (wv - 4) >> 1, tj = (wv - 4) & 1;
      v16bf a = load_a(shAt, CC, ti * 16, 0, lane);
      v16bf b = load_b(shU, DVS, 0, tj * 16, lane);
      oacc = wmma_bf16(a, b, oacc);
      int mh = (lane >> 4) * 8, n = lane & 15;
      float* op = out + (seq_base + l0 + ti * 16 + mh) * (size_t)DVT +
                  (size_t)sl * DVS + tj * 16 + n;
#pragma unroll
      for (int r = 0; r < 8; ++r) op[(size_t)r * DVT] = oacc[r];
    } else {
      float g = shGm;
      for (int i = tid; i < DK * DVS; i += 128) shS[i] *= g;
    }
    __syncthreads();

    // ================= stage G: S += k^T u_i  (f32 accumulate) ================
    for (int t = wv; t < 32; t += 8) {
      int mi = t >> 1, nj = t & 1;
      v16bf a = load_a(shKt, CC, mi * 16, 0, lane);    // A = k^T tile (direct)
      v16bf b = load_b(shU, DVS, 0, nj * 16, lane);
      v8f cc = load_c(shS, DVS, mi * 16, nj * 16, lane);
      cc = wmma_bf16(a, b, cc);
      int mh = (lane >> 4) * 8, n = lane & 15;
#pragma unroll
      for (int r = 0; r < 8; ++r) {
        int idx = (mi * 16 + mh + r) * DVS + nj * 16 + n;
        shS[idx]  = cc[r];
        shSb[idx] = f2bf(cc[r]);
      }
    }
    __syncthreads();
  }

  // ---- write final state slice: outS[bh, dk, sl*DVS + dv] ----
  for (int i = tid; i < DK * DVS; i += 256) {
    int m = i / DVS, d = i % DVS;
    outS[((size_t)bh * DK + m) * DVT + (size_t)sl * DVS + d] = shS[i];
  }
}

extern "C" void kernel_launch(void* const* d_in, const int* in_sizes, int n_in,
                              void* d_out, int out_size, void* d_ws,
                              size_t ws_size, hipStream_t stream) {
  const float* q     = (const float*)d_in[0];
  const float* k     = (const float*)d_in[1];
  const float* v     = (const float*)d_in[2];
  const float* beta  = (const float*)d_in[3];
  const float* gamma = (const float*)d_in[4];
  // d_in[5] = chunk_size (scalar, fixed at 32 in this implementation)

  float* out  = (float*)d_out;
  float* outS = out + (size_t)in_sizes[2];  // out region has v's element count

  int bh = in_sizes[3] / LSEQ;              // b*h = 16
  dim3 grid(bh * NSLICE), block(256);
  hipLaunchKernelGGL(deltanet_kernel, grid, block, 0, stream, q, k, v, beta,
                     gamma, out, outS);
}
